// NNUE_4071628997253
// MI455X (gfx1250) — matrix operand order repaired and checked by reading, not compiled, not run on hardware
//
#include <hip/hip_runtime.h>
#include <stdint.h>

// ---------------------------------------------------------------------------
// NNUE forward for MI455X (gfx1250, wave32, WMMA + Tensor Data Mover).
//   K1: W_ft f32 -> bf16 (RNE), zero-padded 520->528 rows (33 N-tiles of 16).
//   K2: FT GEMM (8192x22528 {0,1}) x W_ft^T via v_wmma_f32_16x16x32_bf16.
//       A tiles (32 x 256 f32) DMA'd global->LDS by the TDM (double buffered,
//       TENSORcnt-tracked), converted f32->bf16 in-register (exact for 0/1).
//       32 M-rows per WG so every B fragment (L2-resident W_ft bf16) feeds
//       two WMMAs -> halved L2 traffic per FLOP, one load-wait per 2 wmmas.
//   K3: per-row MLP tail, one wave32 per batch row, selected bucket only.
// Workspace: Wb (23,789,568 B) + wp (17,039,360 B) + bp (17,039,360 B) ~ 58MB.
// ---------------------------------------------------------------------------

#define B_ROWS     8192
#define NFEAT      22528
#define HIDDEN     512
#define NOUT       520        // HIDDEN + 8 psqt buckets
#define NPAD       528        // padded to 33 tiles of 16
#define KC         256        // K chunk DMA'd into LDS
#define NCHUNK     (NFEAT / KC)          // 88
#define MROWS      32         // batch rows per workgroup
#define ROW_DW     260        // 256 dwords + 4 pad dwords (TDM pad feature)

typedef __attribute__((ext_vector_type(16))) __bf16 v16bf;
typedef __attribute__((ext_vector_type(8)))  float  v8f;
typedef __attribute__((ext_vector_type(4)))  unsigned int u32x4;
typedef __attribute__((ext_vector_type(8)))  int i32x8;
typedef __attribute__((ext_vector_type(4)))  int i32x4;

union FragBF { v16bf v; uint4 q[2]; unsigned u[8]; };

__device__ __forceinline__ unsigned short f32_to_bf16_rne(float f) {
  unsigned u = __float_as_uint(f);
  u += 0x7FFFu + ((u >> 16) & 1u);
  return (unsigned short)(u >> 16);
}

// pack two f32 into a dword of 2 bf16 by truncation (exact for 0.0 / 1.0)
__device__ __forceinline__ unsigned pk(float lo, float hi) {
  return (__float_as_uint(hi) & 0xFFFF0000u) | (__float_as_uint(lo) >> 16);
}

__device__ __forceinline__ float clamp01(float x) {
  return fminf(fmaxf(x, 0.0f), 1.0f);
}

// ---------------------------------------------------------------------------
// TDM: DMA a 32-row x KC-col f32 tile (row stride NFEAT) from global into LDS,
// padding the LDS destination by 4 dwords every 256 dwords (row stride 260).
// D# layout per CDNA5 ISA (group0 / group1 bitfields).
// clang-23 / therock-10.0 exposes the 6-arg builtin form.
// ---------------------------------------------------------------------------
__device__ __forceinline__ void tdm_load_tile(unsigned long long gaddr,
                                              unsigned lds_off) {
  u32x4 g0;
  g0[0] = 1u;                                    // count=1, user mode
  g0[1] = lds_off;                               // LDS byte address
  g0[2] = (unsigned)gaddr;                       // global addr [31:0]
  g0[3] = (unsigned)(gaddr >> 32) | (2u << 30);  // addr[56:32] | type=2
  i32x8 g1;
  g1[0] = (int)((2u << 16)        // data_size = 4B
              | (1u << 20)        // pad_enable
              | (7u << 22)        // pad_interval = 256 dwords
              | (3u << 25));      // pad_amount   = 4 dwords
  g1[1] = (int)((NFEAT & 0xFFFFu) << 16);            // tensor_dim0[15:0]
  g1[2] = (int)((NFEAT >> 16) | ((unsigned)MROWS << 16)); // dim0 hi | tensor_dim1
  g1[3] = (int)(KC << 16);                           // dim1 hi=0 | tile_dim0=KC
  g1[4] = MROWS;                                     // tile_dim1, tile_dim2=0
  g1[5] = (int)NFEAT;                                // tensor_dim0_stride[31:0]
  g1[6] = 0;                                         // stride hi / dim1_stride
  g1[7] = 0;
  i32x4 z4 = {0, 0, 0, 0};
  i32x8 z8 = {0, 0, 0, 0, 0, 0, 0, 0};
  __builtin_amdgcn_tensor_load_to_lds(g0, g1, z4, z4, z8, 0);
}

// ---------------------------------------------------------------------------
// Kernel 1: W_ft (520 x 22528 f32, row-major) -> bf16, zero-pad to 528 rows.
// Row-major bf16 W_ft is directly the WMMA B-fragment byte layout.
// ---------------------------------------------------------------------------
__global__ __launch_bounds__(256) void prep_wft_bf16(
    const float* __restrict__ W_ft, unsigned short* __restrict__ Wb) {
  const int col = blockIdx.x * 256 + threadIdx.x;
  const int row = blockIdx.y;
  const size_t idx = (size_t)row * NFEAT + col;
  unsigned short v = 0;
  if (row < NOUT) v = f32_to_bf16_rne(W_ft[idx]);
  Wb[idx] = v;
}

// ---------------------------------------------------------------------------
// Kernel 2: FT GEMM. grid = (256 M-tiles of 32 rows, 2 {white,black}).
// Wave w owns N-tiles 4w..4w+3 (all n < 512, unconditional) for BOTH 16-row
// halves; wave 0 also owns tile 32 (psqt) behind a scalar branch.
// ---------------------------------------------------------------------------
__global__ __launch_bounds__(256) void ft_gemm_bf16(
    const float* __restrict__ wfeat, const float* __restrict__ bfeat,
    const unsigned short* __restrict__ Wb, const float* __restrict__ b_ft,
    float* __restrict__ wp, float* __restrict__ bp) {
  __shared__ __align__(16) float sBuf[2][MROWS * ROW_DW];   // 2 x 33.3 KB

  const int tid  = threadIdx.x;
  const int lane = tid & 31;                  // wave32
  const int hi   = lane >> 4;                 // half-wave select
  const int wave = tid >> 5;
  const int wvid = __builtin_amdgcn_readfirstlane(tid) >> 5;  // scalar wave id
  const int mtile = blockIdx.x;

  const float* __restrict__ A = blockIdx.y ? bfeat : wfeat;
  float* __restrict__ Out     = blockIdx.y ? bp : wp;

  // B fragment base pointers: W_ft row n, K offset hi*8 (8 bf16 at k, 8 at k+16).
  const unsigned short* Bp[4];
#pragma unroll
  for (int i = 0; i < 4; ++i) {
    int n = (4 * wave + i) * 16 + (lane & 15);
    Bp[i] = Wb + (size_t)n * NFEAT + hi * 8;
  }
  const unsigned short* Bp4 =
      Wb + (size_t)(32 * 16 + (lane & 15)) * NFEAT + hi * 8;   // psqt tile

  v8f accL[4] = {};   // batch rows 0..15 of the M tile
  v8f accH[4] = {};   // batch rows 16..31
  v8f acc4L = {};
  v8f acc4H = {};

  const unsigned lds0 = (unsigned)(uintptr_t)&sBuf[0][0];
  const unsigned lds1 = (unsigned)(uintptr_t)&sBuf[1][0];
  const unsigned long long gbase =
      (unsigned long long)(uintptr_t)(A + (size_t)(mtile * MROWS) * NFEAT);

  // A-fragment read base in LDS (f32): row m = lane&15, K offset hi*8
  const int aOff = (lane & 15) * ROW_DW + hi * 8;

  if (wvid == 0) tdm_load_tile(gbase, lds0);            // prologue DMA

  for (int c = 0; c < NCHUNK; ++c) {
    __builtin_amdgcn_s_wait_tensorcnt(0);   // wave0: chunk c landed in LDS
    __syncthreads();                        // all waves: data ready, prev reads done
    if (wvid == 0 && c + 1 < NCHUNK)        // overlap DMA of c+1 with compute of c
      tdm_load_tile(gbase + (size_t)(c + 1) * KC * 4, ((c + 1) & 1) ? lds1 : lds0);

    const float* __restrict__ buf = &sBuf[c & 1][0];
    const float* aBaseL = buf + aOff;
    const float* aBaseH = buf + aOff + 16 * ROW_DW;
    const size_t kofs = (size_t)c * KC;

#pragma unroll
    for (int ks = 0; ks < 8; ++ks) {
      // two A fragments (rows 0-15 / 16-31): f32 LDS -> bf16 (exact for 0/1)
      FragBF aL, aH;
      {
        const float4 f0 = *(const float4*)(aBaseL + ks * 32);
        const float4 f1 = *(const float4*)(aBaseL + ks * 32 + 4);
        const float4 f2 = *(const float4*)(aBaseL + ks * 32 + 16);
        const float4 f3 = *(const float4*)(aBaseL + ks * 32 + 20);
        aL.u[0] = pk(f0.x, f0.y); aL.u[1] = pk(f0.z, f0.w);
        aL.u[2] = pk(f1.x, f1.y); aL.u[3] = pk(f1.z, f1.w);
        aL.u[4] = pk(f2.x, f2.y); aL.u[5] = pk(f2.z, f2.w);
        aL.u[6] = pk(f3.x, f3.y); aL.u[7] = pk(f3.z, f3.w);
      }
      {
        const float4 f0 = *(const float4*)(aBaseH + ks * 32);
        const float4 f1 = *(const float4*)(aBaseH + ks * 32 + 4);
        const float4 f2 = *(const float4*)(aBaseH + ks * 32 + 16);
        const float4 f3 = *(const float4*)(aBaseH + ks * 32 + 20);
        aH.u[0] = pk(f0.x, f0.y); aH.u[1] = pk(f0.z, f0.w);
        aH.u[2] = pk(f1.x, f1.y); aH.u[3] = pk(f1.z, f1.w);
        aH.u[4] = pk(f2.x, f2.y); aH.u[5] = pk(f2.z, f2.w);
        aH.u[6] = pk(f3.x, f3.y); aH.u[7] = pk(f3.z, f3.w);
      }

      const size_t ko = kofs + ks * 32;
#pragma unroll
      for (int i = 0; i < 4; ++i) {   // each B fragment feeds two WMMAs
        FragBF b;
        b.q[0] = *(const uint4*)(Bp[i] + ko);
        b.q[1] = *(const uint4*)(Bp[i] + ko + 16);
        accL[i] = __builtin_amdgcn_wmma_f32_16x16x32_bf16(
            false, aL.v, false, b.v, (short)0, accL[i], false, false);
        accH[i] = __builtin_amdgcn_wmma_f32_16x16x32_bf16(
            false, aH.v, false, b.v, (short)0, accH[i], false, false);
      }
      if (wvid == 0) {              // scalar branch: psqt tile, wave 0 only
        FragBF b4;
        b4.q[0] = *(const uint4*)(Bp4 + ko);
        b4.q[1] = *(const uint4*)(Bp4 + ko + 16);
        acc4L = __builtin_amdgcn_wmma_f32_16x16x32_bf16(
            false, aL.v, false, b4.v, (short)0, acc4L, false, false);
        acc4H = __builtin_amdgcn_wmma_f32_16x16x32_bf16(
            false, aH.v, false, b4.v, (short)0, acc4H, false, false);
      }
    }
  }

  // epilogue: +b_ft, store f32. C layout: VGPR r -> M = r + hi*8, N = lane&15.
  const int mbase = mtile * MROWS;
#pragma unroll
  for (int i = 0; i < 4; ++i) {
    const int n = (4 * wave + i) * 16 + (lane & 15);   // always < 512
    const float bias = b_ft[n];
#pragma unroll
    for (int r = 0; r < 8; ++r) {
      const int m = mbase + r + hi * 8;
      Out[(size_t)m * NOUT + n]        = accL[i][r] + bias;
      Out[(size_t)(m + 16) * NOUT + n] = accH[i][r] + bias;
    }
  }
  if (wvid == 0) {
    const int n = 32 * 16 + (lane & 15);               // 512..527
    if (n < NOUT) {
      const float bias = b_ft[n];
#pragma unroll
      for (int r = 0; r < 8; ++r) {
        const int m = mbase + r + hi * 8;
        Out[(size_t)m * NOUT + n]        = acc4L[r] + bias;
        Out[(size_t)(m + 16) * NOUT + n] = acc4H[r] + bias;
      }
    }
  }
}

// ---------------------------------------------------------------------------
// Kernel 3: MLP tail. One wave32 per batch row; lane = output index (32 wide).
// Computes only the selected bucket (reference computes all 8 then selects).
// ---------------------------------------------------------------------------
__global__ __launch_bounds__(256) void mlp_tail(
    const float* __restrict__ wp, const float* __restrict__ bp,
    const float* __restrict__ stm, const int* __restrict__ buckets,
    const float* __restrict__ W1, const float* __restrict__ b1,
    const float* __restrict__ W2, const float* __restrict__ b2,
    const float* __restrict__ W3, const float* __restrict__ b3,
    float* __restrict__ out) {
  __shared__ float sx[8][1024];
  const int wave = threadIdx.x >> 5;
  const int lane = threadIdx.x & 31;
  const int row  = blockIdx.x * 8 + wave;

  const float s = stm[row];                 // exactly 0.0 or 1.0
  const int   k = buckets[row];
  const float* wrow = wp + (size_t)row * NOUT;
  const float* brow = bp + (size_t)row * NOUT;
  const bool flip = (s >= 0.5f);
  const float* firstp  = flip ? brow : wrow;
  const float* secondp = flip ? wrow : brow;

  for (int j = lane; j < HIDDEN; j += 32) {
    sx[wave][j]          = clamp01(firstp[j]);
    sx[wave][HIDDEN + j] = clamp01(secondp[j]);
  }
  __syncthreads();

  // layer 1: 1024 -> 32, lane o streams W1[k][o][:] (L2-resident, 1MB total)
  float a1 = b1[k * 32 + lane];
  const float4* xv = (const float4*)&sx[wave][0];
  const float4* wv = (const float4*)(W1 + ((size_t)k * 32 + lane) * 1024);
#pragma unroll 8
  for (int d = 0; d < 256; ++d) {
    float4 x4 = xv[d], w4 = wv[d];
    a1 += x4.x * w4.x + x4.y * w4.y + x4.z * w4.z + x4.w * w4.w;
  }
  float h1 = clamp01(a1);

  // layer 2: 32 -> 32 via cross-lane shuffle (wave32)
  float a2 = b2[k * 32 + lane];
  const float* w2p = W2 + ((size_t)k * 32 + lane) * 32;
#pragma unroll
  for (int i = 0; i < 32; ++i) a2 += __shfl(h1, i) * w2p[i];
  float h2 = clamp01(a2);

  // layer 3: 32 -> 1 wave reduction
  float t = h2 * W3[(size_t)k * 32 + lane];
#pragma unroll
  for (int off = 16; off > 0; off >>= 1) t += __shfl_xor(t, off);

  if (lane == 0) {
    float psqt = (wrow[HIDDEN + k] - brow[HIDDEN + k]) * (0.5f - s);
    out[row] = (t + b3[k] + psqt) * 400.0f;
  }
}

// ---------------------------------------------------------------------------
extern "C" void kernel_launch(void* const* d_in, const int* in_sizes, int n_in,
                              void* d_out, int out_size, void* d_ws, size_t ws_size,
                              hipStream_t stream) {
  (void)in_sizes; (void)n_in; (void)out_size; (void)ws_size;
  const float* wf    = (const float*)d_in[0];
  const float* bfeat = (const float*)d_in[1];
  const float* stm   = (const float*)d_in[2];
  const int*   bkt   = (const int*)d_in[3];
  const float* W_ft  = (const float*)d_in[4];
  const float* b_ft  = (const float*)d_in[5];
  const float* W1    = (const float*)d_in[6];
  const float* b1    = (const float*)d_in[7];
  const float* W2    = (const float*)d_in[8];
  const float* b2    = (const float*)d_in[9];
  const float* W3    = (const float*)d_in[10];
  const float* b3    = (const float*)d_in[11];

  char* ws = (char*)d_ws;
  unsigned short* Wb = (unsigned short*)ws;                 // 23,789,568 B
  float* wp = (float*)(ws + (size_t)NPAD * NFEAT * 2);      // 17,039,360 B
  float* bp = wp + (size_t)B_ROWS * NOUT;                   // 17,039,360 B

  prep_wft_bf16<<<dim3(NFEAT / 256, NPAD), 256, 0, stream>>>(W_ft, Wb);
  ft_gemm_bf16<<<dim3(B_ROWS / MROWS, 2), 256, 0, stream>>>(wf, bfeat, Wb, b_ft, wp, bp);
  mlp_tail<<<dim3(B_ROWS / 8), 256, 0, stream>>>(wp, bp, stm, bkt,
                                                 W1, b1, W2, b2, W3, b3,
                                                 (float*)d_out);
}
